// GKHDA_50938312131105
// MI455X (gfx1250) — compile-verified, matrix operand order checked by reading, not run on hardware
//
#include <hip/hip_runtime.h>
#include <hip/hip_bf16.h>

// ---------------- problem constants (match reference) ----------------
#define N_NODES 4096
#define F_INF   512
#define DHID    64
#define NHEADS  8
#define NCLASS  16
#define ALPHA   0.2f
#define BETA    0.9f
#define NEGBIG  (-1e30f)
#define LOG2E   1.4426950408889634f

typedef __attribute__((ext_vector_type(16))) __bf16 v16bf;
typedef __attribute__((ext_vector_type(8)))  __bf16 v8bf;
typedef __attribute__((ext_vector_type(8)))  float  v8f;
typedef __attribute__((ext_vector_type(4)))  float  v4f;

__device__ __forceinline__ float lrelu(float v) { return v > 0.0f ? v : ALPHA * v; }
__device__ __forceinline__ float eluf(float v)  { return v > 0.0f ? v : __expf(v) - 1.0f; }
__device__ __forceinline__ float shflx(float v, int m) { return __shfl_xor(v, m, 32); }
__device__ __forceinline__ float shflb(float v, int s) { return __shfl(v, s, 32); }

// ---------------------------------------------------------------------
// Kpack: one-time repack of W [H,512,64] and W_out [512,16] (f32, row-major)
// into bf16 ISA B-matrix lane records (32B/lane), so the GEMM inner loops
// do coalesced 32B loads instead of stride-64 b32 gathers.
// 128 waves for W + 16 waves for W_out.
// ---------------------------------------------------------------------
__global__ __launch_bounds__(256) void gat_pack(
    const float* __restrict__ W, const float* __restrict__ Wout,
    __bf16* __restrict__ w_b, __bf16* __restrict__ wout_b)
{
  const int wid  = (blockIdx.x * blockDim.x + threadIdx.x) >> 5;
  const int lane = threadIdx.x & 31;
  const int hl   = lane >> 4;
  const int ln   = lane & 15;

  if (wid < NHEADS * (F_INF / 32)) {
    const int head = wid / (F_INF / 32);
    const int kb   = wid % (F_INF / 32);
    const float* Wh = W + (size_t)head * F_INF * DHID + (size_t)(kb * 32 + 8 * hl) * DHID;
#pragma unroll
    for (int t = 0; t < 4; ++t) {
      const float* wp = Wh + 16 * t + ln;
      v16bf B;
#pragma unroll
      for (int i = 0; i < 8; ++i) {
        B[i]     = (__bf16)wp[i * DHID];
        B[8 + i] = (__bf16)wp[(16 + i) * DHID];
      }
      size_t rec = ((size_t)(head * (F_INF / 32) + kb) * 4 + t) * 32 + lane;
      *(v16bf*)(w_b + rec * 16) = B;
    }
  } else if (wid < NHEADS * (F_INF / 32) + (NHEADS * DHID / 32)) {
    const int kb = wid - NHEADS * (F_INF / 32);
    const float* wp = Wout + (size_t)(kb * 32 + 8 * hl) * NCLASS + ln;
    v16bf B;
#pragma unroll
    for (int i = 0; i < 8; ++i) {
      B[i]     = (__bf16)wp[i * NCLASS];
      B[8 + i] = (__bf16)wp[(16 + i) * NCLASS];
    }
    *(v16bf*)(wout_b + ((size_t)kb * 32 + lane) * 16) = B;
  }
}

// ---------------------------------------------------------------------
// K0: Wh[h] = x @ W[h] via v_wmma_f32_16x16x32_bf16 (packed-B weights).
//     Stores Wh pre-swizzled into B layout (wh_b) for the attention kernel,
//     and f1 = (Wh.a1)*log2e, f2 = (Wh.a2)*log2e  (log2-domain scores so the
//     softmax exp becomes a raw v_exp_f32; valid since lrelu is positively
//     homogeneous and softmax is invariant under base change).
// One wave per (head, 16-row tile).
// ---------------------------------------------------------------------
__global__ __launch_bounds__(256) void gat_k0(
    const float* __restrict__ x, const __bf16* __restrict__ w_b,
    const float* __restrict__ a1, const float* __restrict__ a2,
    __bf16* __restrict__ wh_b, float* __restrict__ f1, float* __restrict__ f2)
{
  const int wid  = (blockIdx.x * blockDim.x + threadIdx.x) >> 5;
  const int lane = threadIdx.x & 31;
  const int hl   = lane >> 4;      // wave half (ISA A/B/C layout split)
  const int ln   = lane & 15;
  const int head = wid / (N_NODES / 16);
  const int m0   = (wid % (N_NODES / 16)) * 16;

  v8f acc[4];
#pragma unroll
  for (int t = 0; t < 4; ++t) acc[t] = (v8f)0.0f;

  const float* xr = x + (size_t)(m0 + ln) * F_INF;

  for (int k = 0; k < F_INF; k += 32) {
    // A tile (16x32): lane ln holds row m0+ln, K split per ISA 16-bit A layout
    v4f alo0 = *(const v4f*)(xr + k + 8 * hl);
    v4f alo1 = *(const v4f*)(xr + k + 8 * hl + 4);
    v4f ahi0 = *(const v4f*)(xr + k + 16 + 8 * hl);
    v4f ahi1 = *(const v4f*)(xr + k + 16 + 8 * hl + 4);
    v16bf A;
#pragma unroll
    for (int i = 0; i < 4; ++i) {
      A[i]      = (__bf16)alo0[i];
      A[4 + i]  = (__bf16)alo1[i];
      A[8 + i]  = (__bf16)ahi0[i];
      A[12 + i] = (__bf16)ahi1[i];
    }
    const __bf16* wb =
        w_b + (((size_t)head * (F_INF / 32) + (k >> 5)) * 4 * 32 + lane) * 16;
#pragma unroll
    for (int t = 0; t < 4; ++t) {
      v16bf B = *(const v16bf*)(wb + (size_t)t * 32 * 16);
      acc[t] = __builtin_amdgcn_wmma_f32_16x16x32_bf16(
          false, A, false, B, (short)0, acc[t], false, false);
    }
  }

  // C-layout -> packed-B-layout store (identity per-lane: element j=half*8+v)
  const int kb   = m0 >> 5;
  const int half = (m0 >> 4) & 1;
#pragma unroll
  for (int t = 0; t < 4; ++t) {
    v8bf o;
#pragma unroll
    for (int v = 0; v < 8; ++v) o[v] = (__bf16)acc[t][v];
    size_t rec = ((size_t)(head * (N_NODES / 32) + kb) * 4 + t) * 32 + lane;
    *(v8bf*)(wh_b + rec * 16 + half * 8) = o;
  }

  // f1/f2: per-lane partials over its column, then 16-lane tree reduction
  float s1[8], s2[8];
#pragma unroll
  for (int v = 0; v < 8; ++v) { s1[v] = 0.f; s2[v] = 0.f; }
#pragma unroll
  for (int t = 0; t < 4; ++t) {
    float a1v = a1[head * DHID + 16 * t + ln];
    float a2v = a2[head * DHID + 16 * t + ln];
#pragma unroll
    for (int v = 0; v < 8; ++v) { s1[v] += acc[t][v] * a1v; s2[v] += acc[t][v] * a2v; }
  }
#pragma unroll
  for (int v = 0; v < 8; ++v)
    for (int m = 1; m < 16; m <<= 1) { s1[v] += shflx(s1[v], m); s2[v] += shflx(s2[v], m); }
  if (ln == 0) {
#pragma unroll
    for (int v = 0; v < 8; ++v) {
      int r = m0 + v + 8 * hl;
      f1[head * N_NODES + r] = s1[v] * LOG2E;   // log2-domain scores
      f2[head * N_NODES + r] = s2[v] * LOG2E;
    }
  }
}

// ---------------------------------------------------------------------
// Online-softmax hop update (log2-domain scores -> raw v_exp_f32).
// e[16]  : scores in A-layout order.  blo/bhi : 8 mask bytes each.
// ms/ls  : running max / running sum for this lane's row (dup'd in lane^16).
// acc    : C-layout accumulators (ntiles x 8).  B : packed Wh tiles.
// ---------------------------------------------------------------------
__device__ __forceinline__ void hop_update(
    const float* e, unsigned long long blo, unsigned long long bhi,
    float& ms, float& ls, const v16bf* B, v8f* acc, int ntiles, int hl)
{
  bool on[16];
  float tm = NEGBIG;
#pragma unroll
  for (int i = 0; i < 16; ++i) {
    unsigned long long bits = (i < 8) ? blo : bhi;
    on[i] = ((bits >> (8 * (i & 7))) & 0xffULL) != 0ULL;
    tm = fmaxf(tm, on[i] ? e[i] : NEGBIG);
  }
  tm = fmaxf(tm, shflx(tm, 16));              // combine lane pair (row dup)
  float mn   = fmaxf(ms, tm);
  float corr = __builtin_amdgcn_exp2f(ms - mn);  // ms==mn==NEGBIG -> 1, ls stays 0
  float ps = 0.f;
  v16bf P;
#pragma unroll
  for (int i = 0; i < 16; ++i) {
    float pe = on[i] ? __builtin_amdgcn_exp2f(e[i] - mn) : 0.f;  // masked -> exact 0
    ps += pe;
    P[i] = (__bf16)pe;
  }
  ps += shflx(ps, 16);
  ls = ls * corr + ps;
  ms = mn;
  // broadcast per-row rescale into C layout (row v+8*hl lives in lane v+8*hl)
  float cc[8];
#pragma unroll
  for (int v = 0; v < 8; ++v) cc[v] = shflb(corr, v + 8 * hl);
  for (int t = 0; t < ntiles; ++t) {
#pragma unroll
    for (int v = 0; v < 8; ++v) acc[t][v] *= cc[v];
    acc[t] = __builtin_amdgcn_wmma_f32_16x16x32_bf16(
        false, P, false, B[t], (short)0, acc[t], false, false);
  }
}

// ---------------------------------------------------------------------
// K1: fused 2-hop masked flash attention per head:
//     h = elu( att0@Wh + BETA * att1@Wh ), scores never materialized.
//     Output written bf16 into x2 concat layout [N, H*D].
// ---------------------------------------------------------------------
__global__ __launch_bounds__(256) void gat_k1(
    const unsigned char* __restrict__ masks, const float* __restrict__ f1,
    const float* __restrict__ f2, const __bf16* __restrict__ wh_b,
    __bf16* __restrict__ x2b)
{
  const int wid  = (blockIdx.x * blockDim.x + threadIdx.x) >> 5;
  const int lane = threadIdx.x & 31;
  const int hl   = lane >> 4;
  const int ln   = lane & 15;
  const int head = wid / (N_NODES / 16);
  const int m0   = (wid % (N_NODES / 16)) * 16;
  const int row  = m0 + ln;

  const float  f1r = f1[head * N_NODES + row];
  const float* f2p = f2 + head * N_NODES;
  const unsigned char* mk0 = masks + (size_t)row * N_NODES;
  const unsigned char* mk1 = mk0 + (size_t)N_NODES * N_NODES;

  v8f acc0[4], acc1[4];
#pragma unroll
  for (int t = 0; t < 4; ++t) { acc0[t] = (v8f)0.f; acc1[t] = (v8f)0.f; }
  float m0s = NEGBIG, m1s = NEGBIG, l0s = 0.f, l1s = 0.f;

  for (int kb = 0; kb < N_NODES; kb += 32) {
    const int olo = kb + 8 * hl, ohi = kb + 16 + 8 * hl;
    // scores e_ij = lrelu(f1_i + f2_j) (log2 domain) in A-layout order
    v4f flo0 = *(const v4f*)(f2p + olo);
    v4f flo1 = *(const v4f*)(f2p + olo + 4);
    v4f fhi0 = *(const v4f*)(f2p + ohi);
    v4f fhi1 = *(const v4f*)(f2p + ohi + 4);
    float e[16];
#pragma unroll
    for (int i = 0; i < 4; ++i) {
      e[i]      = lrelu(f1r + flo0[i]);
      e[4 + i]  = lrelu(f1r + flo1[i]);
      e[8 + i]  = lrelu(f1r + fhi0[i]);
      e[12 + i] = lrelu(f1r + fhi1[i]);
    }
    // packed bool mask bytes (L2-resident), 8 per u64
    unsigned long long ma_lo = *(const unsigned long long*)(mk0 + olo);
    unsigned long long ma_hi = *(const unsigned long long*)(mk0 + ohi);
    unsigned long long mb_lo = *(const unsigned long long*)(mk1 + olo);
    unsigned long long mb_hi = *(const unsigned long long*)(mk1 + ohi);
    __builtin_prefetch(mk0 + olo + 256, 0, 0);   // global_prefetch_b8
    __builtin_prefetch(mk1 + olo + 256, 0, 0);

    // packed Wh B tiles: 32B/lane contiguous, shared by both hops
    const __bf16* wb =
        wh_b + (((size_t)head * (N_NODES / 32) + (kb >> 5)) * 4 * 32 + lane) * 16;
    v16bf B[4];
#pragma unroll
    for (int t = 0; t < 4; ++t) B[t] = *(const v16bf*)(wb + (size_t)t * 32 * 16);

    hop_update(e, ma_lo, ma_hi, m0s, l0s, B, acc0, 4, hl);
    hop_update(e, mb_lo, mb_hi, m1s, l1s, B, acc1, 4, hl);
  }

  // finalize: hop decay, softmax normalization, elu, concat store
  float i0 = 1.0f / l0s;
  float i1 = BETA / l1s;
  float c0[8], c1[8];
#pragma unroll
  for (int v = 0; v < 8; ++v) { c0[v] = shflb(i0, v + 8 * hl); c1[v] = shflb(i1, v + 8 * hl); }
#pragma unroll
  for (int t = 0; t < 4; ++t)
#pragma unroll
    for (int v = 0; v < 8; ++v) {
      float val = eluf(acc0[t][v] * c0[v] + acc1[t][v] * c1[v]);
      x2b[(size_t)(m0 + v + 8 * hl) * (NHEADS * DHID) + head * DHID + 16 * t + ln] =
          (__bf16)val;
    }
}

// ---------------------------------------------------------------------
// K2: Who = x2 @ W_out (one 16-col WMMA tile, packed-B weights),
//     pack Who into B layout, g1/g2 = (Who.a_out)*log2e.
// ---------------------------------------------------------------------
__global__ __launch_bounds__(256) void gat_k2(
    const __bf16* __restrict__ x2b, const __bf16* __restrict__ wout_b,
    const float* __restrict__ ao1, const float* __restrict__ ao2,
    __bf16* __restrict__ who_b, float* __restrict__ g1, float* __restrict__ g2)
{
  const int wid  = (blockIdx.x * blockDim.x + threadIdx.x) >> 5;
  const int lane = threadIdx.x & 31;
  const int hl   = lane >> 4;
  const int ln   = lane & 15;
  const int m0   = wid * 16;

  v8f acc = (v8f)0.f;
  const __bf16* xr = x2b + (size_t)(m0 + ln) * (NHEADS * DHID);
  for (int k = 0; k < NHEADS * DHID; k += 32) {
    v8bf lo = *(const v8bf*)(xr + k + 8 * hl);
    v8bf hi = *(const v8bf*)(xr + k + 16 + 8 * hl);
    v16bf A;
#pragma unroll
    for (int i = 0; i < 8; ++i) { A[i] = lo[i]; A[8 + i] = hi[i]; }
    v16bf B = *(const v16bf*)(wout_b + (((size_t)(k >> 5)) * 32 + lane) * 16);
    acc = __builtin_amdgcn_wmma_f32_16x16x32_bf16(
        false, A, false, B, (short)0, acc, false, false);
  }

  {   // pack Who into B layout for K3
    v8bf o;
#pragma unroll
    for (int v = 0; v < 8; ++v) o[v] = (__bf16)acc[v];
    size_t rec = (size_t)((m0 >> 5) * 32 + lane);
    *(v8bf*)(who_b + rec * 16 + ((m0 >> 4) & 1) * 8) = o;
  }

  float a1v = ao1[ln], a2v = ao2[ln];
  float s1[8], s2[8];
#pragma unroll
  for (int v = 0; v < 8; ++v) { s1[v] = acc[v] * a1v; s2[v] = acc[v] * a2v; }
#pragma unroll
  for (int v = 0; v < 8; ++v)
    for (int m = 1; m < 16; m <<= 1) { s1[v] += shflx(s1[v], m); s2[v] += shflx(s2[v], m); }
  if (ln == 0) {
#pragma unroll
    for (int v = 0; v < 8; ++v) {
      int r = m0 + v + 8 * hl;
      g1[r] = s1[v] * LOG2E;    // log2-domain scores
      g2[r] = s2[v] * LOG2E;
    }
  }
}

// ---------------------------------------------------------------------
// K3: output 2-hop masked flash attention (C=16) + elu + log_softmax.
// ---------------------------------------------------------------------
__global__ __launch_bounds__(256) void gat_k3(
    const unsigned char* __restrict__ masks, const float* __restrict__ g1,
    const float* __restrict__ g2, const __bf16* __restrict__ who_b,
    float* __restrict__ out)
{
  const int wid  = (blockIdx.x * blockDim.x + threadIdx.x) >> 5;
  const int lane = threadIdx.x & 31;
  const int hl   = lane >> 4;
  const int ln   = lane & 15;
  const int m0   = wid * 16;
  const int row  = m0 + ln;

  const float g1r = g1[row];
  const unsigned char* mk0 = masks + (size_t)row * N_NODES;
  const unsigned char* mk1 = mk0 + (size_t)N_NODES * N_NODES;

  v8f acc0[1], acc1[1];
  acc0[0] = (v8f)0.f;
  acc1[0] = (v8f)0.f;
  float m0s = NEGBIG, m1s = NEGBIG, l0s = 0.f, l1s = 0.f;

  for (int kb = 0; kb < N_NODES; kb += 32) {
    const int olo = kb + 8 * hl, ohi = kb + 16 + 8 * hl;
    v4f flo0 = *(const v4f*)(g2 + olo);
    v4f flo1 = *(const v4f*)(g2 + olo + 4);
    v4f fhi0 = *(const v4f*)(g2 + ohi);
    v4f fhi1 = *(const v4f*)(g2 + ohi + 4);
    float e[16];
#pragma unroll
    for (int i = 0; i < 4; ++i) {
      e[i]      = lrelu(g1r + flo0[i]);
      e[4 + i]  = lrelu(g1r + flo1[i]);
      e[8 + i]  = lrelu(g1r + fhi0[i]);
      e[12 + i] = lrelu(g1r + fhi1[i]);
    }
    unsigned long long ma_lo = *(const unsigned long long*)(mk0 + olo);
    unsigned long long ma_hi = *(const unsigned long long*)(mk0 + ohi);
    unsigned long long mb_lo = *(const unsigned long long*)(mk1 + olo);
    unsigned long long mb_hi = *(const unsigned long long*)(mk1 + ohi);
    __builtin_prefetch(mk0 + olo + 256, 0, 0);
    __builtin_prefetch(mk1 + olo + 256, 0, 0);

    v16bf B[1];
    B[0] = *(const v16bf*)(who_b + ((size_t)(kb >> 5) * 32 + lane) * 16);

    hop_update(e, ma_lo, ma_hi, m0s, l0s, B, acc0, 1, hl);
    hop_update(e, mb_lo, mb_hi, m1s, l1s, B, acc1, 1, hl);
  }

  float i0 = 1.0f / l0s;
  float i1 = BETA / l1s;
  float c0[8], c1[8];
#pragma unroll
  for (int v = 0; v < 8; ++v) { c0[v] = shflb(i0, v + 8 * hl); c1[v] = shflb(i1, v + 8 * hl); }
  float val[8];
#pragma unroll
  for (int v = 0; v < 8; ++v) val[v] = eluf(acc0[0][v] * c0[v] + acc1[0][v] * c1[v]);

  // log_softmax over the 16 classes (spread across the 16-lane group)
#pragma unroll
  for (int v = 0; v < 8; ++v) {
    float r = val[v];
    for (int m = 1; m < 16; m <<= 1) r = fmaxf(r, shflx(r, m));
    float se = __expf(val[v] - r);
    for (int m = 1; m < 16; m <<= 1) se += shflx(se, m);
    out[(size_t)(m0 + v + 8 * hl) * NCLASS + ln] = val[v] - r - __logf(se);
  }
}

// ---------------------------------------------------------------------
extern "C" void kernel_launch(void* const* d_in, const int* in_sizes, int n_in,
                              void* d_out, int out_size, void* d_ws, size_t ws_size,
                              hipStream_t stream) {
  (void)in_sizes; (void)n_in; (void)out_size; (void)ws_size;
  const float*         x     = (const float*)d_in[0];
  const unsigned char* masks = (const unsigned char*)d_in[1];   // jnp.bool_
  const float*         W     = (const float*)d_in[2];
  const float*         a1    = (const float*)d_in[3];
  const float*         a2    = (const float*)d_in[4];
  const float*         Wout  = (const float*)d_in[5];
  const float*         ao1   = (const float*)d_in[6];
  const float*         ao2   = (const float*)d_in[7];
  float*               out   = (float*)d_out;

  // workspace carve-out (~9.4 MB), all regions fully overwritten before read
  char*  ws  = (char*)d_ws;
  size_t off = 0;
  __bf16* wh_b   = (__bf16*)(ws + off); off += (size_t)NHEADS * (N_NODES / 32) * 4 * 32 * 16 * 2;
  float*  f1     = (float*)(ws + off);  off += (size_t)NHEADS * N_NODES * 4;
  float*  f2     = (float*)(ws + off);  off += (size_t)NHEADS * N_NODES * 4;
  __bf16* x2b    = (__bf16*)(ws + off); off += (size_t)N_NODES * NHEADS * DHID * 2;
  __bf16* who_b  = (__bf16*)(ws + off); off += (size_t)(N_NODES / 32) * 32 * 16 * 2;
  float*  g1     = (float*)(ws + off);  off += (size_t)N_NODES * 4;
  float*  g2     = (float*)(ws + off);  off += (size_t)N_NODES * 4;
  __bf16* w_b    = (__bf16*)(ws + off); off += (size_t)NHEADS * (F_INF / 32) * 4 * 32 * 16 * 2;
  __bf16* wout_b = (__bf16*)(ws + off); off += (size_t)(NHEADS * DHID / 32) * 32 * 16 * 2;

  const int waves_pack = NHEADS * (F_INF / 32) + (NHEADS * DHID / 32);  // 144
  const int waves_l1   = NHEADS * (N_NODES / 16);                      // 2048
  const int waves_l2   = N_NODES / 16;                                 // 256
  dim3 blk(256);
  gat_pack<<<dim3(waves_pack * 32 / 256), blk, 0, stream>>>(W, Wout, w_b, wout_b);
  gat_k0<<<dim3(waves_l1 * 32 / 256), blk, 0, stream>>>(x, w_b, a1, a2, wh_b, f1, f2);
  gat_k1<<<dim3(waves_l1 * 32 / 256), blk, 0, stream>>>(masks, f1, f2, wh_b, x2b);
  gat_k2<<<dim3(waves_l2 * 32 / 256), blk, 0, stream>>>(x2b, wout_b, ao1, ao2, who_b, g1, g2);
  gat_k3<<<dim3(waves_l2 * 32 / 256), blk, 0, stream>>>(masks, g1, g2, who_b, out);
}
